// ESN_46961172415181
// MI455X (gfx1250) — matrix operand order repaired and checked by reading, not compiled, not run on hardware
//
#include <hip/hip_runtime.h>
#include <hip/hip_bf16.h>
#include <math.h>

// ---------------------------------------------------------------------------
// ESN on MI455X (gfx1250, wave32, WMMA).
//   U = input @ W_in^T          : bf16 WMMA GEMM (16x16x32), written to d_out
//   x_t = erf(U_t + W_res x_km1)/sqrt(N) : persistent kernel, W_res held
//         entirely in VGPRs (64 VGPRs/lane of bf16 A-tiles), matvec done as
//         WMMA with x in column 0 of B, grid barrier per timestep.
// ---------------------------------------------------------------------------

typedef __attribute__((ext_vector_type(16))) __bf16 v16bf;
typedef __attribute__((ext_vector_type(8)))  __bf16 v8bf;
typedef __attribute__((ext_vector_type(8)))  float  v8f;

#define N_RES   2048
#define N_IN    128
#define SEQ_T   2048
#define NWG     128      // workgroups in scan kernel (one 16-row tile each)
#define WAVES   8        // waves per scan workgroup
#define KCHUNKS 8        // 32-wide k-chunks per wave: 8 waves * 8 * 32 = 2048

#define INV_SQRT_N 0.02209708691207961f   // 1/sqrt(2048)

union ATile { v16bf v; v8bf h[2]; };

// ---- workspace layout (bytes) ----
// [0,64)                : grid-barrier counter
// [64, 64+8192)         : xstate, 2 x 2048 bf16 (double buffered)
// [8320, +512KB)        : input  as bf16 (2048 x 128)
// [+512KB, +1MB)        : W_in   as bf16 (2048 x 128)
// [+1MB,  +9MB)         : W_res packed bf16 in WMMA A-tile lane order
#define XOFF   64
#define INOFF  8320
#define WINOFF (8320 + 524288)
#define WPOFF  (8320 + 2*524288)

// --- convert input & W_in to bf16, reset barrier counter -------------------
__global__ void esn_cvt_inputs(const float* __restrict__ inp,
                               const float* __restrict__ win,
                               __bf16* __restrict__ inbf,
                               __bf16* __restrict__ winbf,
                               int* __restrict__ cnt) {
  int i = blockIdx.x * blockDim.x + threadIdx.x;
  if (i == 0) *cnt = 0;
  if (i < N_RES * N_IN) {
    inbf[i]  = (__bf16)inp[i];
    winbf[i] = (__bf16)win[i];
  }
}

// --- pack W_res (fp32 row-major) into bf16 WMMA A-tile lane order ----------
// tile (mTile 0..127, kTile 0..63) = 16x32 elements = 512 bf16, lane-major:
// lane l holds 16 contiguous bf16 (32B) in contiguous K order (K = 16*(l/16) + j).
__global__ void esn_pack_wres(const float* __restrict__ wres,
                              __bf16* __restrict__ wp) {
  int p = blockIdx.x * blockDim.x + threadIdx.x;   // 0 .. 2048*2048-1
  int tile   = p >> 9;
  int within = p & 511;
  int lane = within >> 4;
  int j    = within & 15;
  int mTile = tile >> 6;       // 64 k-tiles per row-tile
  int kTile = tile & 63;
  int M  = lane & 15;
  int kh = lane >> 4;
  int K  = kh * 16 + j;
  int row = mTile * 16 + M;
  int col = kTile * 32 + K;
  wp[p] = (__bf16)wres[(size_t)row * N_RES + col];
}

// --- U = SCALE_IN * input @ W_in^T  (written into d_out) -------------------
__global__ __launch_bounds__(256)
void esn_gemm_u(const __bf16* __restrict__ inbf,
                const __bf16* __restrict__ winbf,
                float* __restrict__ U) {
  int wave = blockIdx.x * (blockDim.x >> 5) + (threadIdx.x >> 5);
  int lane = threadIdx.x & 31;
  int tTile = wave >> 7;       // 128 n-tiles per t-tile row
  int nTile = wave & 127;
  int M  = lane & 15;          // A row within tile
  int cn = lane & 15;          // B column within tile
  int kh = lane >> 4;

  const __bf16* arow = inbf  + (size_t)(tTile * 16 + M)  * N_IN;
  const __bf16* brow = winbf + (size_t)(nTile * 16 + cn) * N_IN;

  v8f acc = {};
#pragma unroll
  for (int dk = 0; dk < N_IN; dk += 32) {
    ATile a, b;
    a.v    = *(const v16bf*)(arow + dk + kh * 16);      // K = kh*16 + 0..15
    b.v    = *(const v16bf*)(brow + dk + kh * 16);      // K = kh*16 + 0..15
    acc = __builtin_amdgcn_wmma_f32_16x16x32_bf16(
        false, a.v, false, b.v, (short)0, acc, false, false);
  }

  int col   = nTile * 16 + (lane & 15);
  int rbase = tTile * 16 + ((lane >> 4) << 3);
#pragma unroll
  for (int r = 0; r < 8; ++r)
    U[(size_t)(rbase + r) * N_RES + col] = 1.0f /*SCALE_IN*/ * acc[r];
}

// --- x0 = erf(U[0]) / sqrt(N), seed state buffer 0 -------------------------
__global__ void esn_x0(float* __restrict__ out, __bf16* __restrict__ xstate) {
  int n = blockIdx.x * blockDim.x + threadIdx.x;
  if (n < N_RES) {
    float x = erff(out[n]) * INV_SQRT_N;
    out[n] = x;
    xstate[n] = (__bf16)x;
  }
}

// --- persistent recurrent scan ---------------------------------------------
// WG m owns output rows [16m, 16m+16). Wave w owns k-chunks w*8 .. w*8+7,
// whose W_res A-tiles are hoisted into 64 VGPRs/lane for the whole scan.
__global__ __launch_bounds__(256)
void esn_scan(const __bf16* __restrict__ wp,
              float* __restrict__ out,
              __bf16* __restrict__ xstate,
              int* __restrict__ cnt) {
  __shared__ float red[WAVES][16];
  const int tid  = threadIdx.x;
  const int w    = tid >> 5;
  const int lane = tid & 31;
  const int m    = blockIdx.x;

  // one-time: W_res tiles -> registers (perfectly coalesced 32B/lane loads)
  v16bf a[KCHUNKS];
  const __bf16* base = wp + ((size_t)m * 64 + (size_t)w * KCHUNKS) * 512 + lane * 16;
#pragma unroll
  for (int i = 0; i < KCHUNKS; ++i)
    a[i] = *(const v16bf*)(base + i * 512);

  const bool bl = (lane == 0) || (lane == 16);  // lanes holding B column 0
  const int  xlo = (lane >> 4) << 4;            // 0 or 16: K offset in chunk

  for (int t = 1; t < SEQ_T; ++t) {
    const __bf16* xr = xstate + (size_t)((t - 1) & 1) * N_RES;
    v8f acc = {};
#pragma unroll
    for (int i = 0; i < KCHUNKS; ++i) {
      int k0 = (w * KCHUNKS + i) * 32;
      v16bf b = {};
      if (bl) b = *(const v16bf*)(xr + k0 + xlo);   // x segment -> B col 0
      acc = __builtin_amdgcn_wmma_f32_16x16x32_bf16(
          false, a[i], false, b, (short)0, acc, false, false);
    }
    // D column 0: lane 0 holds M=0..7 in acc[0..7], lane 16 holds M=8..15
    if (lane == 0) {
#pragma unroll
      for (int r = 0; r < 8; ++r) red[w][r] = acc[r];
    } else if (lane == 16) {
#pragma unroll
      for (int r = 0; r < 8; ++r) red[w][8 + r] = acc[r];
    }
    __syncthreads();

    if (tid < 16) {
      float s = 0.0f;
#pragma unroll
      for (int q = 0; q < WAVES; ++q) s += red[q][tid];
      int row = m * 16 + tid;
      float pre = out[(size_t)t * N_RES + row] + s;    // U[t][row] + W_res.x
      float x = erff(pre) * INV_SQRT_N;
      out[(size_t)t * N_RES + row] = x;
      xstate[(size_t)(t & 1) * N_RES + row] = (__bf16)x;
      __threadfence();                                 // publish x device-wide
    }
    __syncthreads();

    // grid barrier: monotonic counter, release/acquire at agent scope
    if (tid == 0) {
      __hip_atomic_fetch_add(cnt, 1, __ATOMIC_RELEASE, __HIP_MEMORY_SCOPE_AGENT);
      const int target = NWG * t;
      while (__hip_atomic_load(cnt, __ATOMIC_ACQUIRE, __HIP_MEMORY_SCOPE_AGENT) < target)
        __builtin_amdgcn_s_sleep(2);
    }
    __syncthreads();
  }
}

extern "C" void kernel_launch(void* const* d_in, const int* in_sizes, int n_in,
                              void* d_out, int out_size, void* d_ws, size_t ws_size,
                              hipStream_t stream) {
  const float* inp  = (const float*)d_in[0];   // (2048, 128)
  const float* win  = (const float*)d_in[1];   // (2048, 128)
  const float* wres = (const float*)d_in[2];   // (2048, 2048)
  float* out = (float*)d_out;                  // (2048, 2048)

  char* ws = (char*)d_ws;
  int*    cnt    = (int*)ws;
  __bf16* xstate = (__bf16*)(ws + XOFF);
  __bf16* inbf   = (__bf16*)(ws + INOFF);
  __bf16* winbf  = (__bf16*)(ws + WINOFF);
  __bf16* wp     = (__bf16*)(ws + WPOFF);

  esn_cvt_inputs<<<(N_RES * N_IN + 255) / 256, 256, 0, stream>>>(inp, win, inbf, winbf, cnt);
  esn_pack_wres<<<(N_RES * N_RES) / 256, 256, 0, stream>>>(wres, wp);
  esn_gemm_u<<<(128 * 128) / 8, 256, 0, stream>>>(inbf, winbf, out);
  esn_x0<<<(N_RES + 255) / 256, 256, 0, stream>>>(out, xstate);
  esn_scan<<<NWG, 256, 0, stream>>>(wp, out, xstate, cnt);
}
